// HebbianConv2d_1425929142903
// MI455X (gfx1250) — matrix operand order repaired and checked by reading, not compiled, not run on hardware
//
#include <hip/hip_runtime.h>

// ---------------------------------------------------------------------------
// Implicit-GEMM conv2d (3x3 VALID) for gfx1250.
// x: [32,256,56,56] f32, w: [512,256,3,3] f32 -> out: [32,512,54,54] f32
// GEMM: M=C_out=512, N=93312, K=2304, K reordered as k'=(kh*3+kw)*256+ci.
//
// Kernel 1 (one-time): swizzle weights f32 -> bf16 into d_ws in the exact
//   per-chunk fragment layout [chunk][co_sub][lane][e]; a WG's A tile per
//   chunk is then 16 KB CONTIGUOUS in global memory.
// Kernel 2 (main): A tile staged with global_load_async_to_lds_b128
//   (ASYNCcnt, double-buffered LDS) -> zero VGPR traffic / zero converts
//   for A. B staged via division-free im2col gather + v_cvt_pk_bf16_f32.
//   Wave tile 64x64 = 4x4 v_wmma_f32_16x16x32_bf16.
// ---------------------------------------------------------------------------

typedef __attribute__((ext_vector_type(16))) __bf16 v16bf;
typedef __attribute__((ext_vector_type(8)))  __bf16 v8bf;
typedef __attribute__((ext_vector_type(8)))  float  v8f;

#define C_IN   256
#define C_OUT  512
#define H_IN   56
#define W_IN   56
#define H_OUT  54
#define W_OUT  54
#define BATCH  32
#define K_TOTAL   (C_IN * 9)              // 2304
#define PIX_TOTAL (BATCH * H_OUT * W_OUT) // 93312
#define N_CHUNK   (K_TOTAL / 32)          // 72

#define TILE_M 256
#define TILE_N 128
#define TILE_K 32

// ---- pre-pass: w[co][ci][kh][kw] f32 -> bf16 fragment image in d_ws --------
// layout: wf[chunk 0..71][co_sub 0..31][lane 0..31][e 0..15]
__global__ __launch_bounds__(256) void swizzle_weights(
    const float* __restrict__ w, __bf16* __restrict__ wf)
{
  const int idx   = blockIdx.x * 256 + threadIdx.x;   // 0 .. 1,179,647
  const int e     = idx & 15;
  const int ln    = (idx >> 4) & 31;
  const int sub   = (idx >> 9) & 31;
  const int chunk = idx >> 14;
  const int co = (sub << 4) + (ln & 15);
  // A-fragment element -> K offset within chunk (ISA 16-bit A 16x32 layout)
  const int kk = ((e >> 3) << 4) + ((ln >> 4) << 3) +
                 (((e >> 1) & 3) << 1) + (e & 1);
  const int kp   = chunk * 32 + kk;       // reordered K index
  const int ci   = kp & 255;
  const int khkw = kp >> 8;               // 0..8
  wf[idx] = (__bf16)w[co * K_TOTAL + ci * 9 + khkw];
}

// ---- main kernel ------------------------------------------------------------
__global__ __launch_bounds__(256) void conv3x3_wmma_bf16(
    const float* __restrict__ x, const __bf16* __restrict__ wf,
    float* __restrict__ out)
{
  // A: double-buffered [co_sub 0..15][lane][e] = 16 KB x 2 (async-filled)
  // B: [p_sub 0..7][lane][e] = 8 KB (reg-staged)
  __shared__ __align__(32) __bf16 sA[2][16 * 32 * 16];
  __shared__ __align__(32) __bf16 sB[8 * 32 * 16];

  const int tid  = threadIdx.x;
  const int wave = tid >> 5;
  const int lane = tid & 31;

  const int m_tile = blockIdx.x * TILE_M;
  const int p_tile = blockIdx.y * TILE_N;

  const int wm = (wave & 3) << 6;   // 4 waves along M, 64 co each
  const int wn = (wave >> 2) << 6;  // 2 waves along N, 64 px each

  // ---- B staging runs: 2 runs of 8 consecutive k' per thread ----
  int b_base[2];
  int b_loff[2];
  #pragma unroll
  for (int t = 0; t < 2; ++t) {
    const int r   = tid + t * 256;
    const int sub = r >> 6;
    const int ln  = (r >> 1) & 31;
    const int h   = r & 1;
    const int p   = p_tile + (sub << 4) + (ln & 15);
    const int b   = p / (H_OUT * W_OUT);
    const int pr  = p - b * (H_OUT * W_OUT);
    const int ho  = pr / W_OUT;
    const int wo  = pr - ho * W_OUT;
    const int kk0 = ((ln >> 4) << 3) + (h << 4);
    b_base[t] = b * (C_IN * H_IN * W_IN) + ho * W_IN + wo
              + kk0 * (H_IN * W_IN);
    b_loff[t] = (((sub << 5) + ln) << 4) + (h << 3);
  }

  v8f acc[4][4] = {};
  float rb[2][8];

  // A tile for (chunk, m_tile) is 16 KB contiguous in wf.
  auto issue_asyncA = [&](int chunk, int buf) {
    const char* g = (const char*)(wf + (size_t)chunk * (32 * 512)
                                     + ((m_tile >> 4) << 9));
    const unsigned lbase = (unsigned)(size_t)(void*)&sA[buf][0] + tid * 16;
    #pragma unroll
    for (int it = 0; it < 4; ++it) {
      const unsigned lds = lbase + it * 4096;
      const void* gp = g + tid * 16 + it * 4096;
      asm volatile("global_load_async_to_lds_b128 %0, %1, off"
                   :: "v"(lds), "v"(gp) : "memory");
    }
  };

  auto prefetchB = [&](int kc) {
    const int ci0  = kc & 255;
    const int khkw = kc >> 8;                 // uniform SALU tap decode
    const int kh   = khkw / 3;
    const int kw   = khkw - kh * 3;
    const float* xp = x + ci0 * (H_IN * W_IN) + kh * W_IN + kw;
    #pragma unroll
    for (int t = 0; t < 2; ++t)
      #pragma unroll
      for (int j = 0; j < 8; ++j)
        rb[t][j] = xp[b_base[t] + j * (H_IN * W_IN)];   // imm-offset loads
  };

  issue_asyncA(0, 0);
  prefetchB(0);

  for (int kc = 0; kc < K_TOTAL; kc += TILE_K) {
    const int buf = (kc >> 5) & 1;

    // B regs -> LDS (packed cvt, one ds_store_b128 per run)
    #pragma unroll
    for (int t = 0; t < 2; ++t) {
      v8bf pb;
      #pragma unroll
      for (int j = 0; j < 8; ++j) pb[j] = (__bf16)rb[t][j];
      *(v8bf*)(sB + b_loff[t]) = pb;
    }
    // A(kc) async copies into sA[buf] complete:
    asm volatile("s_wait_asynccnt 0" ::: "memory");
    __syncthreads();

    // Fill the other A buffer + next B regs while WMMAs run.
    if (kc + TILE_K < K_TOTAL) {
      issue_asyncA((kc >> 5) + 1, buf ^ 1);
      prefetchB(kc + TILE_K);
    }

    const v16bf* pA = (const v16bf*)sA[buf];
    const v16bf* pB = (const v16bf*)sB;
    v16bf afr[4], bfr[4];
    #pragma unroll
    for (int i = 0; i < 4; ++i) afr[i] = pA[(((wm >> 4) + i) << 5) + lane];
    #pragma unroll
    for (int j = 0; j < 4; ++j) bfr[j] = pB[(((wn >> 4) + j) << 5) + lane];

    #pragma unroll
    for (int i = 0; i < 4; ++i)
      #pragma unroll
      for (int j = 0; j < 4; ++j)
        acc[i][j] = __builtin_amdgcn_wmma_f32_16x16x32_bf16(
            false, afr[i], false, bfr[j], (short)0, acc[i][j], false, false);
    __syncthreads();
  }

  // ---- store (C/D f32 layout: lane half selects M = r or r+8; N = lane&15) --
  const int nloc  = lane & 15;
  const int mhalf = (lane >> 4) << 3;
  #pragma unroll
  for (int j = 0; j < 4; ++j) {
    const int p  = p_tile + wn + (j << 4) + nloc;
    const int b  = p / (H_OUT * W_OUT);
    const int pr = p - b * (H_OUT * W_OUT);
    const int ho = pr / W_OUT;
    const int wo = pr - ho * W_OUT;
    #pragma unroll
    for (int i = 0; i < 4; ++i) {
      const int co0  = m_tile + wm + (i << 4) + mhalf;
      const int base = ((b * C_OUT + co0) * H_OUT + ho) * W_OUT + wo;
      #pragma unroll
      for (int r = 0; r < 8; ++r)
        out[base + r * (H_OUT * W_OUT)] = acc[i][j][r];
    }
  }
}

extern "C" void kernel_launch(void* const* d_in, const int* in_sizes, int n_in,
                              void* d_out, int out_size, void* d_ws, size_t ws_size,
                              hipStream_t stream) {
  const float* x = (const float*)d_in[0];   // [32,256,56,56]
  const float* w = (const float*)d_in[1];   // [512,256,3,3]
  float* out     = (float*)d_out;           // [32,512,54,54]
  __bf16* wf     = (__bf16*)d_ws;           // 72*32*32*16 bf16 = 2.25 MB

  // one-time weight swizzle (re-run every call: deterministic, inputs untouched)
  swizzle_weights<<<dim3(N_CHUNK * 32 * 32 * 16 / 256), dim3(256), 0, stream>>>(w, wf);

  dim3 grid(C_OUT / TILE_M, PIX_TOTAL / TILE_N);  // 2 x 729
  conv3x3_wmma_bf16<<<grid, dim3(256), 0, stream>>>(x, wf, out);
}
